// QuantumGeometricAttention_88957362634984
// MI455X (gfx1250) — matrix operand order repaired and checked by reading, not compiled, not run on hardware
//
#include <hip/hip_runtime.h>
#include <stdint.h>

// ---------------- types ----------------
typedef __attribute__((ext_vector_type(16))) __bf16 v16bf;
typedef __attribute__((ext_vector_type(2)))  __bf16 bf16x2;
typedef __attribute__((ext_vector_type(8)))  float  v8f;

union Frag {
    uint32_t u[8];
    uint2    u2[4];
    uint4    u4[2];
    v16bf    v;
};

__device__ __forceinline__ uint16_t f2bf(float x) {
    union { __bf16 b; uint16_t u; } c; c.b = (__bf16)x;   // native v_cvt_*_bf16_f32
    return c.u;
}
__device__ __forceinline__ uint32_t pk2(float lo, float hi) {
    union { bf16x2 v; uint32_t u; } c;
    c.v[0] = (__bf16)lo; c.v[1] = (__bf16)hi;             // v_cvt_pk_bf16_f32
    return c.u;
}
__device__ __forceinline__ v8f wmma_bf16(const Frag& a, const Frag& b, v8f c) {
    return __builtin_amdgcn_wmma_f32_16x16x32_bf16(false, a.v, false, b.v,
                                                   (short)0, c, false, false);
}

// ---------------- problem constants ----------------
#define HID   1024
#define HEADS 8
#define HD    128
#define MD    8
#define NLAY  3
#define BB    8
#define SS    1024
#define MTOT  (BB * SS)          // 8192 rows

// ================= weight convert+transpose: f32 [K][N] -> bf16 [N][K] ==========
__global__ void cvt_transpose(const float* __restrict__ in, uint16_t* __restrict__ out,
                              int K, int N) {
    size_t i = (size_t)blockIdx.x * 256 + threadIdx.x;
    size_t total = (size_t)K * N;
    if (i >= total) return;
    int n = (int)(i / K);
    int k = (int)(i - (size_t)n * K);
    out[i] = f2bf(in[(size_t)k * N + n]);
}

// ================= generic bf16 WMMA GEMM (double-buffered + L2 prefetch) =======
// C[M][N] = A[M][K](f32, cvt on the fly) * BT[N][K](bf16) + bias[N]
// 128x128 tile / 256 threads (8 waves). Wave w: rows w*16..w*16+15, all 128 cols.
__global__ __launch_bounds__(256) void gemm_bf16(
    const float* __restrict__ A, const uint32_t* __restrict__ BT,
    const float* __restrict__ bias, float* __restrict__ C,
    int M, int N, int K) {
    __shared__ uint32_t ldsA[2][2048];   // [128 rows][16 dw]  (32 k, bf16 pairs)
    __shared__ uint32_t ldsB[2][2048];   // [128 n   ][16 dw]
    const int t = threadIdx.x;
    const int w = t >> 5, lane = t & 31, ln = lane & 15, hi = lane >> 4;
    const int Nb = blockIdx.x * 128, Mb = blockIdx.y * 128;
    const int Kd = K >> 1;

    v8f acc[8];
    #pragma unroll
    for (int n = 0; n < 8; ++n)
        #pragma unroll
        for (int r = 0; r < 8; ++r) acc[n][r] = 0.f;

    // staging addresses for this thread (uint4 granularity: 512 uint4 / tile)
    const int srow = t >> 2, sq = t & 3;            // j stride adds 64 rows

    // prologue: stage tile k0=0 into buffer 0
    #pragma unroll
    for (int j = 0; j < 2; ++j) {
        int row = srow + 64 * j;
        const float4* ga = (const float4*)(A + (size_t)(Mb + row) * K + 8 * sq);
        float4 f0 = ga[0], f1 = ga[1];
        uint4 wv; wv.x = pk2(f0.x, f0.y); wv.y = pk2(f0.z, f0.w);
        wv.z = pk2(f1.x, f1.y); wv.w = pk2(f1.z, f1.w);
        *(uint4*)&ldsA[0][(size_t)row * 16 + sq * 4] = wv;
        *(uint4*)&ldsB[0][(size_t)row * 16 + sq * 4] =
            *(const uint4*)(BT + (size_t)(Nb + row) * Kd + 4 * sq);
    }
    __syncthreads();

    int buf = 0;
    for (int k0 = 0; k0 < K; k0 += 32) {
        const int nk = k0 + 32;
        // ---- issue next tile's global loads early (overlap with WMMA) ----
        float4 ra0[2], ra1[2]; uint4 rb[2];
        if (nk < K) {
            #pragma unroll
            for (int j = 0; j < 2; ++j) {
                int row = srow + 64 * j;
                const float4* ga = (const float4*)(A + (size_t)(Mb + row) * K + nk + 8 * sq);
                ra0[j] = ga[0]; ra1[j] = ga[1];
                rb[j] = *(const uint4*)(BT + (size_t)(Nb + row) * Kd + (nk >> 1) + 4 * sq);
            }
        }
        // ---- deep prefetch (2 tiles ahead) into L2: global_prefetch_b8 ----
        if (k0 + 64 < K) {
            int row = srow;
            __builtin_prefetch(A + (size_t)(Mb + row) * K + (k0 + 64) + 8 * sq, 0, 1);
            __builtin_prefetch(BT + (size_t)(Nb + row) * Kd + ((k0 + 64) >> 1) + 4 * sq, 0, 1);
        }

        // ---- compute on current buffer ----
        Frag fa;                                  // A-fragment (contiguous k = 16*hi + s, same as B)
        const uint32_t* pa = &ldsA[buf][(w * 16 + ln) * 16];
        fa.u4[0] = *(const uint4*)(pa + 8 * hi);
        fa.u4[1] = *(const uint4*)(pa + 8 * hi + 4);

        Frag fb[8];                               // preload all B-fragments
        #pragma unroll
        for (int n = 0; n < 8; ++n) {
            const uint32_t* pb = &ldsB[buf][(n * 16 + ln) * 16 + 8 * hi];
            fb[n].u4[0] = *(const uint4*)(pb);
            fb[n].u4[1] = *(const uint4*)(pb + 4);
        }
        #pragma unroll
        for (int n = 0; n < 8; ++n)               // 8 back-to-back WMMAs
            acc[n] = wmma_bf16(fa, fb[n], acc[n]);

        // ---- commit next tile to the other buffer ----
        if (nk < K) {
            #pragma unroll
            for (int j = 0; j < 2; ++j) {
                int row = srow + 64 * j;
                uint4 wv; wv.x = pk2(ra0[j].x, ra0[j].y); wv.y = pk2(ra0[j].z, ra0[j].w);
                wv.z = pk2(ra1[j].x, ra1[j].y); wv.w = pk2(ra1[j].z, ra1[j].w);
                *(uint4*)&ldsA[buf ^ 1][(size_t)row * 16 + sq * 4] = wv;
                *(uint4*)&ldsB[buf ^ 1][(size_t)row * 16 + sq * 4] = rb[j];
            }
        }
        __syncthreads();
        buf ^= 1;
    }

    #pragma unroll
    for (int n = 0; n < 8; ++n) {
        int col = Nb + n * 16 + ln;
        float bv = bias[col];
        #pragma unroll
        for (int r = 0; r < 8; ++r) {
            int row = Mb + w * 16 + r + 8 * hi;
            C[(size_t)row * N + col] = acc[n][r] + bv;
        }
    }
}

// ================= expmap0 on Q/K, layout shuffle, bf16 convert =================
// one wave per (b,h,s). qf/kf: [BH][S][HD] bf16. vT: [BH][HD][S] bf16.
__global__ __launch_bounds__(256) void prep_qkv(
    const float* __restrict__ Qkv, uint32_t* __restrict__ qf,
    uint32_t* __restrict__ kfp, uint16_t* __restrict__ vT) {
    const int wid  = (blockIdx.x << 3) + (threadIdx.x >> 5);  // 0..65535
    const int lane = threadIdx.x & 31;
    const int b = wid >> 13, h = (wid >> 10) & 7, s = wid & 1023;
    const int m = (b << 10) + s;
    const float* qrow = Qkv + (size_t)m * (3 * HID) + h * HD;
    const float* krow = qrow + HID;
    const float* vrow = qrow + 2 * HID;
    float4 q4 = *(const float4*)(qrow + lane * 4);
    float4 k4 = *(const float4*)(krow + lane * 4);
    float4 v4 = *(const float4*)(vrow + lane * 4);

    float sq = q4.x * q4.x + q4.y * q4.y + q4.z * q4.z + q4.w * q4.w;
    float sk = k4.x * k4.x + k4.y * k4.y + k4.z * k4.z + k4.w * k4.w;
    #pragma unroll
    for (int msk = 1; msk < 32; msk <<= 1) {
        sq += __shfl_xor(sq, msk);
        sk += __shfl_xor(sk, msk);
    }
    float nq = fmaxf(sqrtf(sq), 1e-7f);
    float nk = fmaxf(sqrtf(sk), 1e-7f);
    float tq = tanhf(nq) / nq;
    float tk = tanhf(nk) / nk;

    size_t bhs = (size_t)wid;                 // (b*8+h)*1024+s
    uint32_t* qd = qf  + bhs * (HD / 2) + lane * 2;
    uint32_t* kd = kfp + bhs * (HD / 2) + lane * 2;
    qd[0] = pk2(q4.x * tq, q4.y * tq); qd[1] = pk2(q4.z * tq, q4.w * tq);
    kd[0] = pk2(k4.x * tk, k4.y * tk); kd[1] = pk2(k4.z * tk, k4.w * tk);

    size_t bh = (size_t)(wid >> 10);
    size_t vb = bh * (size_t)(HD * SS) + s;
    int d0 = lane * 4;
    vT[vb + (size_t)(d0 + 0) * SS] = f2bf(v4.x);
    vT[vb + (size_t)(d0 + 1) * SS] = f2bf(v4.y);
    vT[vb + (size_t)(d0 + 2) * SS] = f2bf(v4.z);
    vT[vb + (size_t)(d0 + 3) * SS] = f2bf(v4.w);
}

// ================= flash attention (bf16 WMMA, online softmax, dbl-buffered) ====
// grid: (BH=64, S/128=8). Block 256 = 8 waves, wave handles 16 q rows.
__global__ __launch_bounds__(256) void flash_attn(
    const uint32_t* __restrict__ qf, const uint32_t* __restrict__ kfp,
    const uint32_t* __restrict__ vT, float* __restrict__ O) {
    __shared__ uint32_t ldsK[2][2048];    // [32 keys][64 dw]  (128 d bf16)
    __shared__ uint32_t ldsV[2][2048];    // [128 d  ][16 dw]  (32 keys bf16)
    __shared__ uint32_t ldsP[8 * 256];    // per-wave 16x32 bf16 P tile
    const int t = threadIdx.x, w = t >> 5, lane = t & 31, ln = lane & 15, hi = lane >> 4;
    const int bh = blockIdx.x, b = bh >> 3, h = bh & 7;
    const int qb = blockIdx.y * 128;
    const float isd = 0.08838834764831845f;   // 1/sqrt(128)

    Frag fq[4];                               // 16 rows x 128 d, resident in VGPRs
    {
        const uint32_t* qbase = qf + ((size_t)bh * SS + qb + w * 16 + ln) * (HD / 2);
        #pragma unroll
        for (int ks = 0; ks < 4; ++ks) {
            const uint32_t* p = qbase + ks * 16;
            fq[ks].u4[0] = *(const uint4*)(p + 8 * hi);
            fq[ks].u4[1] = *(const uint4*)(p + 8 * hi + 4);
        }
    }

    v8f oacc[8];
    float mrow[8], lrow[8];
    #pragma unroll
    for (int n = 0; n < 8; ++n)
        #pragma unroll
        for (int r = 0; r < 8; ++r) oacc[n][r] = 0.f;
    #pragma unroll
    for (int r = 0; r < 8; ++r) { mrow[r] = -1e30f; lrow[r] = 0.f; }

    uint16_t* pP = (uint16_t*)&ldsP[w * 256];
    const uint32_t* kbase = kfp + (size_t)bh * SS * (HD / 2);
    const uint32_t* vbase = vT + (size_t)bh * HD * (SS / 2);
    // staging coords (uint4 granularity)
    const int kkey = t >> 4, kc4 = t & 15;    // ldsK: 32 keys x 16 uint4
    const int vd = t >> 2, vc4 = t & 3;       // ldsV: 128 d x 4 uint4 (x2 j)

    // prologue: stage kb=0 into buffer 0
    #pragma unroll
    for (int j = 0; j < 2; ++j) {
        int key = kkey + 16 * j;
        *(uint4*)&ldsK[0][(size_t)key * 64 + kc4 * 4] =
            *(const uint4*)(kbase + (size_t)key * 64 + kc4 * 4);
        int d = vd + 64 * j;
        *(uint4*)&ldsV[0][(size_t)d * 16 + vc4 * 4] =
            *(const uint4*)(vbase + (size_t)d * 512 + vc4 * 4);
    }
    __syncthreads();

    int buf = 0;
    for (int kb = 0; kb < SS; kb += 32) {
        const int nkb = kb + 32;
        // ---- issue next K/V tile global loads early ----
        uint4 rk[2], rv[2];
        if (nkb < SS) {
            #pragma unroll
            for (int j = 0; j < 2; ++j) {
                int key = kkey + 16 * j;
                rk[j] = *(const uint4*)(kbase + (size_t)(nkb + key) * 64 + kc4 * 4);
                int d = vd + 64 * j;
                rv[j] = *(const uint4*)(vbase + (size_t)d * 512 + (nkb >> 1) + vc4 * 4);
            }
        }
        // ---- deep prefetch (2 tiles ahead) into L2 ----
        if (kb + 64 < SS) {
            __builtin_prefetch(kbase + (size_t)(kb + 64 + kkey) * 64 + kc4 * 4, 0, 1);
            __builtin_prefetch(vbase + (size_t)vd * 512 + ((kb + 64) >> 1) + vc4 * 4, 0, 1);
        }

        // ---- scores: 16 q x 32 keys ----
        v8f s0, s1;
        #pragma unroll
        for (int r = 0; r < 8; ++r) { s0[r] = 0.f; s1[r] = 0.f; }
        Frag fk[8];                                // preload all K-fragments
        #pragma unroll
        for (int ks = 0; ks < 4; ++ks) {
            const uint32_t* pk0 = &ldsK[buf][(ln)      * 64 + ks * 16 + 8 * hi];
            const uint32_t* pk1 = &ldsK[buf][(16 + ln) * 64 + ks * 16 + 8 * hi];
            fk[2 * ks].u4[0]     = *(const uint4*)(pk0);
            fk[2 * ks].u4[1]     = *(const uint4*)(pk0 + 4);
            fk[2 * ks + 1].u4[0] = *(const uint4*)(pk1);
            fk[2 * ks + 1].u4[1] = *(const uint4*)(pk1 + 4);
        }
        #pragma unroll
        for (int ks = 0; ks < 4; ++ks) {
            s0 = wmma_bf16(fq[ks], fk[2 * ks],     s0);
            s1 = wmma_bf16(fq[ks], fk[2 * ks + 1], s1);
        }

        // ---- online softmax per row ----
        #pragma unroll
        for (int r = 0; r < 8; ++r) {
            float a = s0[r] * isd, c = s1[r] * isd;
            float mx = fmaxf(a, c);
            #pragma unroll
            for (int msk = 1; msk < 16; msk <<= 1) mx = fmaxf(mx, __shfl_xor(mx, msk));
            float mn = fmaxf(mrow[r], mx);
            float alpha = __expf(mrow[r] - mn);
            mrow[r] = mn;
            float p0 = __expf(a - mn), p1 = __expf(c - mn);
            float rs = p0 + p1;
            #pragma unroll
            for (int msk = 1; msk < 16; msk <<= 1) rs += __shfl_xor(rs, msk);
            lrow[r] = lrow[r] * alpha + rs;
            #pragma unroll
            for (int n = 0; n < 8; ++n) oacc[n][r] *= alpha;
            pP[(r + 8 * hi) * 32 + ln]      = f2bf(p0);   // transpose P via LDS
            pP[(r + 8 * hi) * 32 + 16 + ln] = f2bf(p1);
        }

        Frag fp;                                   // P as A-fragment (16x32)
        const uint32_t* pp = &ldsP[w * 256 + ln * 16];
        fp.u4[0] = *(const uint4*)(pp + 8 * hi);
        fp.u4[1] = *(const uint4*)(pp + 8 * hi + 4);

        Frag fv[8];                                // preload all V-fragments
        #pragma unroll
        for (int n = 0; n < 8; ++n) {
            const uint32_t* pv = &ldsV[buf][(n * 16 + ln) * 16 + 8 * hi];
            fv[n].u4[0] = *(const uint4*)(pv);
            fv[n].u4[1] = *(const uint4*)(pv + 4);
        }
        #pragma unroll
        for (int n = 0; n < 8; ++n)                // O += P @ V  (8 WMMAs)
            oacc[n] = wmma_bf16(fp, fv[n], oacc[n]);

        // ---- commit next K/V tile ----
        if (nkb < SS) {
            #pragma unroll
            for (int j = 0; j < 2; ++j) {
                int key = kkey + 16 * j;
                *(uint4*)&ldsK[buf ^ 1][(size_t)key * 64 + kc4 * 4] = rk[j];
                int d = vd + 64 * j;
                *(uint4*)&ldsV[buf ^ 1][(size_t)d * 16 + vc4 * 4] = rv[j];
            }
        }
        __syncthreads();
        buf ^= 1;
    }

    #pragma unroll
    for (int n = 0; n < 8; ++n)
        #pragma unroll
        for (int r = 0; r < 8; ++r) {
            int row = qb + w * 16 + r + 8 * hi;
            O[((size_t)(b * SS + row)) * HID + h * HD + n * 16 + ln] =
                oacc[n][r] / lrow[r];
        }
}

// ================= MD=8 bottleneck projection + residual =================
__global__ __launch_bounds__(256) void proj_update(
    const float* __restrict__ O, const float* __restrict__ Wp,
    const float* __restrict__ bp, const float* __restrict__ Wa,
    const float* __restrict__ ba, int layer,
    const float* __restrict__ cur_in, float* __restrict__ cur_out) {
    __shared__ float red[8][257];
    __shared__ float m8[8];
    const int t = threadIdx.x;
    const size_t m = blockIdx.x;
    const float* orow = O + m * HID;

    float acc[8];
    #pragma unroll
    for (int j = 0; j < 8; ++j) acc[j] = 0.f;
    #pragma unroll
    for (int i = 0; i < 4; ++i) {
        int k = t + 256 * i;
        float ov = orow[k];
        const float* wr = Wp + (size_t)k * MD;
        #pragma unroll
        for (int j = 0; j < 8; ++j) acc[j] += ov * wr[j];
    }
    #pragma unroll
    for (int j = 0; j < 8; ++j) red[j][t] = acc[j];
    __syncthreads();
    for (int s = 128; s > 0; s >>= 1) {
        if (t < s) {
            #pragma unroll
            for (int j = 0; j < 8; ++j) red[j][t] += red[j][t + s];
        }
        __syncthreads();
    }
    if (t < 8) m8[t] = red[t][0] + bp[t];
    __syncthreads();

    const float* wl = Wa + (size_t)layer * MD * HID;
    const float* bl = ba + (size_t)layer * HID;
    #pragma unroll
    for (int c = 0; c < 4; ++c) {
        int n = t + 256 * c;
        float u = bl[n];
        #pragma unroll
        for (int j = 0; j < 8; ++j) u += m8[j] * wl[j * HID + n];
        cur_out[m * HID + n] = cur_in[m * HID + n] + u;
    }
}

// ================= host orchestration =================
extern "C" void kernel_launch(void* const* d_in, const int* in_sizes, int n_in,
                              void* d_out, int out_size, void* d_ws, size_t ws_size,
                              hipStream_t stream) {
    const float* x      = (const float*)d_in[0];
    const float* W_qkv  = (const float*)d_in[1];
    const float* b_qkv  = (const float*)d_in[2];
    const float* W_pinv = (const float*)d_in[3];
    const float* b_pinv = (const float*)d_in[4];
    const float* W_attn = (const float*)d_in[5];
    const float* b_attn = (const float*)d_in[6];
    const float* W_out  = (const float*)d_in[7];
    const float* b_out  = (const float*)d_in[8];
    float* out = (float*)d_out;

    char* ws = (char*)d_ws;
    const size_t OFF_WQKVT = 0;                               // 3072*1024*2 = 6 MiB
    const size_t OFF_WOUTT = OFF_WQKVT + 6291456;             // 1024*1024*2 = 2 MiB
    const size_t OFF_QF    = OFF_WOUTT + 2097152;             // 16 MiB
    const size_t OFF_KF    = OFF_QF + 16777216;               // 16 MiB
    const size_t OFF_VT    = OFF_KF + 16777216;               // 16 MiB
    const size_t OFF_CUR   = OFF_VT + 16777216;               // 32 MiB
    const size_t OFF_QKV   = OFF_CUR + 33554432;              // 96 MiB (O aliases)

    uint16_t* WqkvT = (uint16_t*)(ws + OFF_WQKVT);
    uint16_t* WoutT = (uint16_t*)(ws + OFF_WOUTT);
    uint32_t* qfb   = (uint32_t*)(ws + OFF_QF);
    uint32_t* kfb   = (uint32_t*)(ws + OFF_KF);
    uint16_t* vTb   = (uint16_t*)(ws + OFF_VT);
    float*    cur   = (float*)   (ws + OFF_CUR);
    float*    Qkv   = (float*)   (ws + OFF_QKV);
    float*    Obuf  = Qkv;    // alias: Qkv dead after prep_qkv

    {
        size_t tot = (size_t)HID * 3 * HID;
        cvt_transpose<<<(unsigned)((tot + 255) / 256), 256, 0, stream>>>(W_qkv, WqkvT, HID, 3 * HID);
    }
    {
        size_t tot = (size_t)HID * HID;
        cvt_transpose<<<(unsigned)((tot + 255) / 256), 256, 0, stream>>>(W_out, WoutT, HID, HID);
    }

    for (int l = 0; l < NLAY; ++l) {
        const float* curA = (l == 0) ? x : cur;
        gemm_bf16<<<dim3(24, 64), 256, 0, stream>>>(curA, (const uint32_t*)WqkvT,
                                                    b_qkv, Qkv, MTOT, 3 * HID, HID);
        prep_qkv<<<8192, 256, 0, stream>>>(Qkv, qfb, kfb, vTb);
        flash_attn<<<dim3(64, 8), 256, 0, stream>>>(qfb, kfb, (const uint32_t*)vTb, Obuf);
        proj_update<<<8192, 256, 0, stream>>>(Obuf, W_pinv, b_pinv, W_attn, b_attn, l,
                                              curA, cur);
    }
    gemm_bf16<<<dim3(8, 64), 256, 0, stream>>>(cur, (const uint32_t*)WoutT,
                                               b_out, out, MTOT, HID, HID);
}
